// RobertaGCN_10531259809996
// MI455X (gfx1250) — matrix operand order repaired and checked by reading, not compiled, not run on hardware
//
#include <hip/hip_runtime.h>
#include <cstdint>
#include <cstddef>

// ---------------- problem constants (match reference) ----------------
constexpr int Bn  = 64;    // batch
constexpr int Sn  = 512;   // sequence
constexpr int Dn  = 768;   // hidden dim
constexpr int H1n = 256;
constexpr int H2n = 128;
constexpr float THRc = 0.1f;

// ---------------- CDNA5 async load-to-LDS (guarded; falls back cleanly) -----
#if defined(__has_builtin)
#  if __has_builtin(__builtin_amdgcn_global_load_async_to_lds_b128) && \
      __has_builtin(__builtin_amdgcn_s_wait_asynccnt)
#    define USE_ASYNC_LDS 1
#  endif
#endif

// The builtin takes pointers to 16-byte int vectors in explicit address spaces:
//   arg0: addrspace(1) int4*  (global source)
//   arg1: addrspace(3) int4*  (LDS destination)
typedef int v4i_ __attribute__((ext_vector_type(4)));
typedef __attribute__((address_space(1))) v4i_ gv4i;
typedef __attribute__((address_space(3))) v4i_ sv4i;

// ---------------- WMMA types ----------------
typedef __attribute__((ext_vector_type(16))) __bf16   bf16x16;
typedef __attribute__((ext_vector_type(8)))  float    f32x8;
typedef __attribute__((ext_vector_type(8)))  uint32_t u32x8;

// fp32 -> bf16 (round to nearest even)
__device__ __forceinline__ unsigned short f2bf(float f) {
  uint32_t u = __float_as_uint(f);
  uint32_t r = u + 0x7FFFu + ((u >> 16) & 1u);
  return (unsigned short)(r >> 16);
}

__device__ __forceinline__ f32x8 wmma_bf16(bf16x16 a, bf16x16 b, f32x8 c) {
  // D = A(16x32 bf16) * B(32x16 bf16) + C(16x16 f32)
  return __builtin_amdgcn_wmma_f32_16x16x32_bf16(
      false, a, false, b, (short)0, c, false, false);
}

// A fragment (16x32, M x K), source row-major with contiguous K (global or LDS).
// ISA layout: lane L<16 -> row M=L, K = {0..7, 16..23}; lane L+16 -> K = {8..15, 24..31}.
__device__ __forceinline__ bf16x16 load_A_frag(const unsigned short* base, int ld) {
  int lane = threadIdx.x & 31;
  int row  = lane & 15;
  int half = lane >> 4;
  const unsigned short* p0 = base + (size_t)row * ld + half * 8;
  uint4 a = *(const uint4*)p0;
  uint4 b = *(const uint4*)(p0 + 16);
  u32x8 w;
  w[0]=a.x; w[1]=a.y; w[2]=a.z; w[3]=a.w;
  w[4]=b.x; w[5]=b.y; w[6]=b.z; w[7]=b.w;
  return __builtin_bit_cast(bf16x16, w);
}

// B fragment (32x16, K x N) where the SOURCE is row-major over (n, k)
// i.e. src[n*ld + k]  (H rows for the gram; transposed weights/activations elsewhere).
__device__ __forceinline__ bf16x16 load_B_frag_kcontig(const unsigned short* __restrict__ base, int ld) {
  int lane = threadIdx.x & 31;
  int col  = lane & 15;
  int koff = (lane >> 4) * 8;
  const unsigned short* p = base + (size_t)col * ld + koff;
  uint4 a = *(const uint4*)p;
  uint4 b = *(const uint4*)(p + 16);
  u32x8 w;
  w[0]=a.x; w[1]=a.y; w[2]=a.z; w[3]=a.w;
  w[4]=b.x; w[5]=b.y; w[6]=b.z; w[7]=b.w;
  return __builtin_bit_cast(bf16x16, w);
}

// Stage ELEMS u16 (multiple of 2048) from global into LDS with 256 threads.
// Uses CDNA5 async load-to-LDS when the builtin exists; plain b128 copies otherwise.
template <int ELEMS>
__device__ __forceinline__ void stage_panel(const unsigned short* __restrict__ g,
                                            unsigned short* l) {
#if defined(USE_ASYNC_LDS)
#pragma unroll
  for (int idx = 0; idx < ELEMS; idx += 2048) {
    int e = idx + threadIdx.x * 8;
    __builtin_amdgcn_global_load_async_to_lds_b128(
        (gv4i*)(g + e), (sv4i*)(l + e), 0, 0);
  }
  __builtin_amdgcn_s_wait_asynccnt(0);
#else
#pragma unroll
  for (int idx = 0; idx < ELEMS; idx += 2048) {
    int e = idx + threadIdx.x * 8;
    *(uint4*)(l + e) = *(const uint4*)(g + e);
  }
#endif
  __syncthreads();
}

// ---------------- kernels ----------------

__global__ void k_cast_bf16(const float* __restrict__ src, unsigned short* __restrict__ dst, int n) {
  int i = blockIdx.x * blockDim.x + threadIdx.x;
  if (i < n) dst[i] = f2bf(src[i]);
}

// src (R x C row-major, f32) -> dst (C x R row-major, bf16)  [small weight matrices]
__global__ void k_cast_bf16_T(const float* __restrict__ src, unsigned short* __restrict__ dst,
                              int R, int C) {
  int i = blockIdx.x * blockDim.x + threadIdx.x;
  if (i < R * C) {
    int r = i / C, c = i - r * C;
    dst[(size_t)c * R + r] = f2bf(src[i]);
  }
}

__global__ void k_zero_f32(float* __restrict__ p, int n) {
  int i = blockIdx.x * blockDim.x + threadIdx.x;
  if (i < n) p[i] = 0.0f;
}

// Fused: logits (WMMA bf16) -> softmax row block -> threshold + self-loop ->
// store A^T (bf16) + column-degree partial sums.
// grid: (S/16, B), block: 256 (8 waves, each wave does 4 16x16 tiles along the row).
__global__ void k_gram_softmax(const unsigned short* __restrict__ Hb,
                               unsigned short* __restrict__ AT,
                               float* __restrict__ deg) {
  __shared__ float att[16][Sn];                 // 32 KB
  __shared__ unsigned short Apanel[16 * Dn];    // 24 KB: shared A rows for all 8 waves
  __shared__ float red[16][16];
  __shared__ float rowmax[16];
  __shared__ float rowsum[16];

  const int b  = blockIdx.y;
  const int i0 = blockIdx.x * 16;
  const int tid  = threadIdx.x;
  const int wave = tid >> 5;
  const int lane = tid & 31;
  const unsigned short* Hbb = Hb + (size_t)b * Sn * Dn;

  stage_panel<16 * Dn>(Hbb + (size_t)i0 * Dn, Apanel);

  f32x8 acc[4] = {};
  // B double-buffer; A comes from LDS (cheap, low latency)
  bf16x16 b_c[4];
#pragma unroll
  for (int t = 0; t < 4; ++t)
    b_c[t] = load_B_frag_kcontig(Hbb + (size_t)((wave * 4 + t) * 16) * Dn, Dn);

#pragma unroll
  for (int kb = 0; kb < Dn; kb += 32) {
    const int nk = kb + 32;
    bf16x16 b_n[4] = {b_c[0], b_c[1], b_c[2], b_c[3]};
    if (nk < Dn) {
#pragma unroll
      for (int t = 0; t < 4; ++t)
        b_n[t] = load_B_frag_kcontig(Hbb + (size_t)((wave * 4 + t) * 16) * Dn + nk, Dn);
    }
    bf16x16 a = load_A_frag(Apanel + kb, Dn);
#pragma unroll
    for (int t = 0; t < 4; ++t) acc[t] = wmma_bf16(a, b_c[t], acc[t]);
#pragma unroll
    for (int t = 0; t < 4; ++t) b_c[t] = b_n[t];
  }

  // C layout: VGPR r, lane L: row = r + 8*(L>>4), col = L&15
  {
    int cm = lane >> 4, cn = lane & 15;
#pragma unroll
    for (int t = 0; t < 4; ++t) {
      int j0 = (wave * 4 + t) * 16;
#pragma unroll
      for (int r = 0; r < 8; ++r) att[r + 8 * cm][j0 + cn] = acc[t][r];
    }
  }
  __syncthreads();

  // softmax over each of the 16 rows (512 values), 16 threads per row
  const int row = tid >> 4, c16 = tid & 15;
  float mx = -3.0e38f;
  for (int j = c16; j < Sn; j += 16) mx = fmaxf(mx, att[row][j]);
  red[row][c16] = mx;
  __syncthreads();
  if (c16 == 0) {
    float m = red[row][0];
#pragma unroll
    for (int k = 1; k < 16; ++k) m = fmaxf(m, red[row][k]);
    rowmax[row] = m;
  }
  __syncthreads();
  float rm = rowmax[row];
  float s = 0.0f;
  for (int j = c16; j < Sn; j += 16) s += __expf(att[row][j] - rm);
  red[row][c16] = s;
  __syncthreads();
  if (c16 == 0) {
    float m = 0.0f;
#pragma unroll
    for (int k = 0; k < 16; ++k) m += red[row][k];
    rowsum[row] = m;
  }
  __syncthreads();
  const float inv = 1.0f / rowsum[row];
  const int gi = i0 + row;
  for (int j = c16; j < Sn; j += 16) {
    float p = __expf(att[row][j] - rm) * inv;
    float v = (p > THRc) ? p : 0.0f;
    if (j == gi) v = (p > THRc) ? p : 1.0f;   // add_remaining_self_loops
    att[row][j] = v;
  }
  __syncthreads();

  // column partial degree + transposed bf16 store (AT[b][j][i] = A[i][j])
#pragma unroll
  for (int pass = 0; pass < 2; ++pass) {
    int jj = tid + pass * 256;
    float colsum = 0.0f;
    uint32_t w[8];
#pragma unroll
    for (int t = 0; t < 8; ++t) {
      float v0 = att[2 * t][jj];
      float v1 = att[2 * t + 1][jj];
      colsum += v0 + v1;
      w[t] = (uint32_t)f2bf(v0) | ((uint32_t)f2bf(v1) << 16);
    }
    atomicAdd(&deg[(size_t)b * Sn + jj], colsum);
    unsigned short* dst = AT + ((size_t)b * Sn + jj) * Sn + i0;
    *(uint4*)dst       = make_uint4(w[0], w[1], w[2], w[3]);
    *(uint4*)(dst + 8) = make_uint4(w[4], w[5], w[6], w[7]);
  }
}

__global__ void k_dis(const float* __restrict__ deg, float* __restrict__ dis, int n) {
  int i = blockIdx.x * blockDim.x + threadIdx.x;
  if (i < n) {
    float d = deg[i];
    dis[i] = (d > 0.0f) ? rsqrtf(d) : 0.0f;
  }
}

// outT[b,n,m] = dis[b,m] * (X[b,m,:] @ Wt[n,:])  stored TRANSPOSED in bf16 so that
// each lane emits one b128 store and the consumer reads K-contiguous.
// grid: (S/16, N/128, B)
template <int K, int N>
__global__ void k_gemm_xw(const unsigned short* __restrict__ X,
                          const unsigned short* __restrict__ Wt,   // [N][K]
                          const float* __restrict__ dis,
                          unsigned short* __restrict__ outT) {    // [B][N][S]
  __shared__ unsigned short Apanel[16 * K];
  const int b    = blockIdx.z;
  const int m0   = blockIdx.x * 16;
  const int wave = threadIdx.x >> 5;
  const int lane = threadIdx.x & 31;
  const int n0   = blockIdx.y * 128 + wave * 16;
  const unsigned short* Wrow = Wt + (size_t)n0 * K;

  stage_panel<16 * K>(X + (size_t)b * Sn * K + (size_t)m0 * K, Apanel);

  f32x8 acc = {};
  bf16x16 b_c = load_B_frag_kcontig(Wrow, K);
#pragma unroll
  for (int kb = 0; kb < K; kb += 32) {
    const int nk = kb + 32;
    bf16x16 b_n = b_c;
    if (nk < K) b_n = load_B_frag_kcontig(Wrow + nk, K);
    bf16x16 a = load_A_frag(Apanel + kb, K);
    acc = wmma_bf16(a, b_c, acc);
    b_c = b_n;
  }
  const int cm = lane >> 4, cn = lane & 15;
  const int mb = m0 + 8 * cm;                 // 8 consecutive rows per lane
  uint32_t w[4];
#pragma unroll
  for (int t = 0; t < 4; ++t) {
    float v0 = acc[2 * t]     * dis[(size_t)b * Sn + mb + 2 * t];
    float v1 = acc[2 * t + 1] * dis[(size_t)b * Sn + mb + 2 * t + 1];
    w[t] = (uint32_t)f2bf(v0) | ((uint32_t)f2bf(v1) << 16);
  }
  *(uint4*)(outT + (size_t)b * Sn * N + (size_t)(n0 + cn) * Sn + mb) =
      make_uint4(w[0], w[1], w[2], w[3]);
}

// y[b,j,n] = relu(dis[b,j] * sum_i A[i,j]*XsT[b,n,i] + bias[n])  (row-major output:
// it is the A operand of the next GEMM).  grid: (S/16, N/128, B)
template <int N>
__global__ void k_agg_relu(const unsigned short* __restrict__ AT,   // [B][S][S]
                           const unsigned short* __restrict__ XsT,  // [B][N][S]
                           const float* __restrict__ dis,
                           const float* __restrict__ bias,
                           unsigned short* __restrict__ out) {      // [B][S][N]
  __shared__ unsigned short Apanel[16 * Sn];   // 16 KB adjacency rows, shared by 8 waves
  const int b    = blockIdx.z;
  const int j0   = blockIdx.x * 16;
  const int wave = threadIdx.x >> 5;
  const int lane = threadIdx.x & 31;
  const int n0   = blockIdx.y * 128 + wave * 16;
  const unsigned short* Brow = XsT + (size_t)b * Sn * N + (size_t)n0 * Sn;

  stage_panel<16 * Sn>(AT + (size_t)b * Sn * Sn + (size_t)j0 * Sn, Apanel);

  f32x8 acc = {};
  bf16x16 b_c = load_B_frag_kcontig(Brow, Sn);
#pragma unroll
  for (int kb = 0; kb < Sn; kb += 32) {
    const int nk = kb + 32;
    bf16x16 b_n = b_c;
    if (nk < Sn) b_n = load_B_frag_kcontig(Brow + nk, Sn);
    bf16x16 a = load_A_frag(Apanel + kb, Sn);
    acc = wmma_bf16(a, b_c, acc);
    b_c = b_n;
  }
  const int cm = lane >> 4, cn = lane & 15;
  const float bval = bias[n0 + cn];
#pragma unroll
  for (int r = 0; r < 8; ++r) {
    int m = j0 + r + 8 * cm;
    float v = acc[r] * dis[(size_t)b * Sn + m] + bval;
    out[(size_t)b * Sn * N + (size_t)m * N + n0 + cn] = f2bf(fmaxf(v, 0.0f));
  }
}

// layer-2 aggregation fused with masked-sum into emb[b,n] (division by mask sum in head).
template <int N>
__global__ void k_agg_masked_sum(const unsigned short* __restrict__ AT,
                                 const unsigned short* __restrict__ XsT,  // [B][N][S]
                                 const float* __restrict__ dis,
                                 const float* __restrict__ bias,
                                 const int* __restrict__ mask,
                                 float* __restrict__ emb) {
  __shared__ unsigned short Apanel[16 * Sn];
  const int b    = blockIdx.z;
  const int j0   = blockIdx.x * 16;
  const int wave = threadIdx.x >> 5;
  const int lane = threadIdx.x & 31;
  const int n0   = blockIdx.y * 128 + wave * 16;
  const unsigned short* Brow = XsT + (size_t)b * Sn * N + (size_t)n0 * Sn;

  stage_panel<16 * Sn>(AT + (size_t)b * Sn * Sn + (size_t)j0 * Sn, Apanel);

  f32x8 acc = {};
  bf16x16 b_c = load_B_frag_kcontig(Brow, Sn);
#pragma unroll
  for (int kb = 0; kb < Sn; kb += 32) {
    const int nk = kb + 32;
    bf16x16 b_n = b_c;
    if (nk < Sn) b_n = load_B_frag_kcontig(Brow + nk, Sn);
    bf16x16 a = load_A_frag(Apanel + kb, Sn);
    acc = wmma_bf16(a, b_c, acc);
    b_c = b_n;
  }
  const int cm = lane >> 4, cn = lane & 15;
  const float bval = bias[n0 + cn];
  float part = 0.0f;
#pragma unroll
  for (int r = 0; r < 8; ++r) {
    int m = j0 + r + 8 * cm;
    float v = acc[r] * dis[(size_t)b * Sn + m] + bval;
    part += v * (float)mask[(size_t)b * Sn + m];
  }
  part += __shfl_xor(part, 16, 32);       // fold the two row-halves of the tile
  if (lane < 16) atomicAdd(&emb[(size_t)b * H2n + n0 + cn], part);
}

// head: out[b] = sigmoid((emb[b,:]/msum[b]) . Wc + bc)
__global__ void k_head(const float* __restrict__ emb,
                       const int* __restrict__ mask,
                       const float* __restrict__ Wc,
                       const float* __restrict__ bc,
                       float* __restrict__ out) {
  int b = threadIdx.x;
  if (b >= Bn) return;
  float msum = 0.0f;
  for (int s = 0; s < Sn; ++s) msum += (float)mask[(size_t)b * Sn + s];
  float inv = 1.0f / msum;
  float z = bc[0];
  for (int h = 0; h < H2n; ++h) z += emb[(size_t)b * H2n + h] * inv * Wc[h];
  out[b] = 1.0f / (1.0f + __expf(-z));
}

// ---------------- launch ----------------
extern "C" void kernel_launch(void* const* d_in, const int* in_sizes, int n_in,
                              void* d_out, int out_size, void* d_ws, size_t ws_size,
                              hipStream_t stream) {
  (void)in_sizes; (void)n_in; (void)out_size; (void)ws_size;
  const float* H   = (const float*)d_in[0];
  const int*   msk = (const int*)  d_in[1];
  const float* W1  = (const float*)d_in[2];
  const float* b1  = (const float*)d_in[3];
  const float* W2  = (const float*)d_in[4];
  const float* b2  = (const float*)d_in[5];
  const float* Wc  = (const float*)d_in[6];
  const float* bc  = (const float*)d_in[7];
  float* out = (float*)d_out;

  // workspace carve-out (~121 MB), all 256B aligned
  char* ws = (char*)d_ws;
  auto carve = [&](size_t bytes) {
    char* p = ws;
    ws += (bytes + 255) & ~(size_t)255;
    return p;
  };
  unsigned short* Hb    = (unsigned short*)carve((size_t)Bn * Sn * Dn * 2);
  unsigned short* W1t   = (unsigned short*)carve((size_t)Dn * H1n * 2);   // [H1][D]
  unsigned short* W2t   = (unsigned short*)carve((size_t)H1n * H2n * 2);  // [H2][H1]
  unsigned short* AT    = (unsigned short*)carve((size_t)Bn * Sn * Sn * 2);
  float*          deg   = (float*)carve((size_t)Bn * Sn * 4);
  float*          dis   = (float*)carve((size_t)Bn * Sn * 4);
  unsigned short* xw1sT = (unsigned short*)carve((size_t)Bn * Sn * H1n * 2); // [B][H1][S]
  unsigned short* y1    = (unsigned short*)carve((size_t)Bn * Sn * H1n * 2); // [B][S][H1]
  unsigned short* xw2sT = (unsigned short*)carve((size_t)Bn * Sn * H2n * 2); // [B][H2][S]
  float*          emb   = (float*)carve((size_t)Bn * H2n * 4);

  // 1) bf16 casts (weights transposed so GEMM B-operands are K-contiguous)
  {
    int n = Bn * Sn * Dn;
    k_cast_bf16<<<(n + 255) / 256, 256, 0, stream>>>(H, Hb, n);
    n = Dn * H1n;
    k_cast_bf16_T<<<(n + 255) / 256, 256, 0, stream>>>(W1, W1t, Dn, H1n);
    n = H1n * H2n;
    k_cast_bf16_T<<<(n + 255) / 256, 256, 0, stream>>>(W2, W2t, H1n, H2n);
  }
  // 2) zero accumulators
  k_zero_f32<<<(Bn * Sn + 255) / 256, 256, 0, stream>>>(deg, Bn * Sn);
  k_zero_f32<<<(Bn * H2n + 255) / 256, 256, 0, stream>>>(emb, Bn * H2n);

  // 3) fused gram + softmax + threshold + self-loop -> AT (bf16) + deg
  k_gram_softmax<<<dim3(Sn / 16, Bn), 256, 0, stream>>>(Hb, AT, deg);

  // 4) dis = rsqrt(deg)
  k_dis<<<(Bn * Sn + 255) / 256, 256, 0, stream>>>(deg, dis, Bn * Sn);

  // 5) layer 1: xw1sT = (dis .* (H @ W1))^T ; y1 = relu(dis .* (A^T @ xw1s) + b1)
  k_gemm_xw<Dn, H1n><<<dim3(Sn / 16, H1n / 128, Bn), 256, 0, stream>>>(Hb, W1t, dis, xw1sT);
  k_agg_relu<H1n><<<dim3(Sn / 16, H1n / 128, Bn), 256, 0, stream>>>(AT, xw1sT, dis, b1, y1);

  // 6) layer 2: xw2sT = (dis .* (y1 @ W2))^T ; emb += mask .* (dis .* (A^T @ xw2s) + b2)
  k_gemm_xw<H1n, H2n><<<dim3(Sn / 16, H2n / 128, Bn), 256, 0, stream>>>(y1, W2t, dis, xw2sT);
  k_agg_masked_sum<H2n><<<dim3(Sn / 16, H2n / 128, Bn), 256, 0, stream>>>(AT, xw2sT, dis, b2, msk, emb);

  // 7) head
  k_head<<<1, 64, 0, stream>>>(emb, msk, Wc, bc, out);
}
